// DGI_25546465477092
// MI455X (gfx1250) — matrix operand order repaired and checked by reading, not compiled, not run on hardware
//
#include <hip/hip_runtime.h>
#include <hip/hip_bf16.h>

// ---------------------------------------------------------------------------
// Types for CDNA5 WMMA (wave32): v16bf A/B fragments, v8f accumulator.
// ---------------------------------------------------------------------------
typedef __bf16 bf16_t;
typedef bf16_t v16bf __attribute__((ext_vector_type(16)));
typedef float  v8f   __attribute__((ext_vector_type(8)));
typedef unsigned int u32x4 __attribute__((ext_vector_type(4)));

union Frag {
    v16bf v;
    u32x4 q[2];
};

// float -> bf16 (round-to-nearest-even), stored as ushort
static __device__ __forceinline__ unsigned short f2bf(float f) {
    unsigned int u = __float_as_uint(f);
    u += 0x7FFFu + ((u >> 16) & 1u);
    return (unsigned short)(u >> 16);
}

// ---------------------------------------------------------------------------
// Degree / normalization: deg starts at 1 (self loop), += 1 per incoming edge,
// then dinv = rsqrt(deg).
// ---------------------------------------------------------------------------
__global__ void dgi_deg_init(float* __restrict__ deg, int n) {
    int i = blockIdx.x * 256 + threadIdx.x;
    if (i < n) deg[i] = 1.0f;
}

__global__ void dgi_deg_count(float* __restrict__ deg, const int* __restrict__ dst, int e) {
    int i = blockIdx.x * 256 + threadIdx.x;
    if (i < e) unsafeAtomicAdd(&deg[dst[i]], 1.0f);
}

__global__ void dgi_deg_fin(float* __restrict__ deg, int n) {
    int i = blockIdx.x * 256 + threadIdx.x;
    if (i < n) deg[i] = __frsqrt_rn(deg[i]);
}

// ---------------------------------------------------------------------------
// Convert weights fp32 [K x 128] -> bf16 pre-packed into WMMA B-fragment
// layout: element (k,n): kbt=k/32, kr=k%32, lane=(kr/16)*16+(n%16), j=kr%16,
// nt=n/16  ->  Bp[((kbt*8+nt)*32+lane)*16 + j]. Each lane's 16 halves are
// contiguous (two b128 loads in the GEMM).
// ---------------------------------------------------------------------------
__global__ void dgi_pack_w(const float* __restrict__ W, unsigned short* __restrict__ Bp, int K) {
    int idx = blockIdx.x * 256 + threadIdx.x;
    if (idx >= K * 128) return;
    int k = idx >> 7, n = idx & 127;
    int kbt = k >> 5, kr = k & 31;
    int lane = ((kr >> 4) << 4) + (n & 15);
    int j = kr & 15;
    int nt = n >> 4;
    int off = (((kbt * 8 + nt) * 32 + lane) << 4) + j;
    Bp[off] = f2bf(W[idx]);
}

// ---------------------------------------------------------------------------
// Convert x fp32 [N x 512] -> bf16 [Npad x 512], optionally gathering rows via
// perm (corrupted encode). Pad rows are zeroed -> GEMM needs no bounds checks.
// ---------------------------------------------------------------------------
__global__ void dgi_conv_x(const float* __restrict__ x, const int* __restrict__ perm,
                           unsigned short* __restrict__ xb, int n, int npad) {
    int idx = blockIdx.x * 256 + threadIdx.x;   // < npad*512 <= ~51.3M, fits int
    int row = idx >> 9;
    int k = idx & 511;
    if (row >= npad) return;
    float v = 0.0f;
    if (row < n) {
        int srow = perm ? perm[row] : row;
        v = x[(size_t)srow * 512 + k];
    }
    xb[idx] = f2bf(v);
}

// ---------------------------------------------------------------------------
// WMMA GEMM: C[Npad x 128] (f32) = A[Npad x K] (bf16) * Bpacked[K x 128] (bf16)
// Block = 256 threads = 8 waves; wave w computes M-tile (blockIdx.x*8+w),
// looping all 8 N-tiles (N=128 total) and K in steps of 32.
// EXEC is all-ones everywhere (no divergence) as WMMA requires.
// ---------------------------------------------------------------------------
__global__ __launch_bounds__(256) void dgi_gemm_bf16(
    const unsigned short* __restrict__ A, const unsigned short* __restrict__ Bp,
    float* __restrict__ C, int K)
{
    const int lane  = threadIdx.x & 31;
    const int wave  = threadIdx.x >> 5;
    const int mtile = blockIdx.x * 8 + wave;
    const int kh    = lane >> 4;      // 0/1: K-half selector for A, K-group for B
    const int l15   = lane & 15;

    const size_t arow = (size_t)(mtile * 16 + l15) * (size_t)K;

    v8f acc[8];
#pragma unroll
    for (int i = 0; i < 8; ++i) {
        v8f z = {0.f, 0.f, 0.f, 0.f, 0.f, 0.f, 0.f, 0.f};
        acc[i] = z;
    }

    for (int kb = 0; kb < K; kb += 32) {
        // A fragment: row M = lane&15, this lane's K values are one contiguous
        // 32-byte run at ushort offset kb+16*kh.
        Frag a;
        const unsigned short* ap = A + arow + kb + 16 * kh;
        a.q[0] = *(const u32x4*)ap;
        a.q[1] = *(const u32x4*)(ap + 8);

        const int kbt = kb >> 5;
#pragma unroll
        for (int nt = 0; nt < 8; ++nt) {
            Frag b;
            const unsigned short* bp = Bp + ((((size_t)kbt * 8 + nt) * 32 + lane) << 4);
            b.q[0] = *(const u32x4*)bp;
            b.q[1] = *(const u32x4*)(bp + 8);
            acc[nt] = __builtin_amdgcn_wmma_f32_16x16x32_bf16(
                false, a.v, false, b.v, (short)0, acc[nt], false, false);
        }
    }

    // C/D layout: lane L, VGPR r -> (M = r + 8*(L/16), N = L%16)
    float* crow = C + (size_t)(mtile * 16 + 8 * kh) * 128;
#pragma unroll
    for (int nt = 0; nt < 8; ++nt) {
        float* cp = crow + nt * 16 + l15;
#pragma unroll
        for (int r = 0; r < 8; ++r) cp[(size_t)r * 128] = acc[nt][r];
    }
}

// ---------------------------------------------------------------------------
// Self-loop init: acc[i,f] = H[i,f] * dinv[i]^2   (also initializes acc)
// ---------------------------------------------------------------------------
__global__ void dgi_selfinit(const float* __restrict__ H, const float* __restrict__ dinv,
                             float* __restrict__ acc, int total) {
    int idx = blockIdx.x * 256 + threadIdx.x;
    if (idx >= total) return;
    float di = dinv[idx >> 7];
    acc[idx] = H[idx] * di * di;
}

// ---------------------------------------------------------------------------
// Edge scatter: one wave per edge; lane handles 4 features (float4 gather,
// 4 hardware f32 atomics). H rows are 512B -> fully coalesced gathers, and
// the whole 51MB feature matrix lives in the 192MB L2.
// ---------------------------------------------------------------------------
__global__ __launch_bounds__(256) void dgi_scatter(
    const float* __restrict__ H, float* __restrict__ acc, const float* __restrict__ dinv,
    const int* __restrict__ src, const int* __restrict__ dst, int e)
{
    int ei = blockIdx.x * 8 + (threadIdx.x >> 5);
    if (ei >= e) return;
    int lane = threadIdx.x & 31;
    int s = src[ei], d = dst[ei];
    float w = dinv[s] * dinv[d];
    float4 v = ((const float4*)(H + (size_t)s * 128))[lane];
    float* p = acc + (size_t)d * 128 + lane * 4;
    unsafeAtomicAdd(p + 0, v.x * w);
    unsafeAtomicAdd(p + 1, v.y * w);
    unsafeAtomicAdd(p + 2, v.z * w);
    unsafeAtomicAdd(p + 3, v.w * w);
}

// +bias, ReLU, -> bf16 (pad rows zeroed for the next GEMM)
__global__ void dgi_finish1(const float* __restrict__ acc, const float* __restrict__ b1,
                            unsigned short* __restrict__ h1b, int n, int npad) {
    int idx = blockIdx.x * 256 + threadIdx.x;
    if (idx >= npad * 128) return;
    int row = idx >> 7;
    float v = 0.0f;
    if (row < n) {
        v = acc[idx] + b1[idx & 127];
        v = v > 0.0f ? v : 0.0f;
    }
    h1b[idx] = f2bf(v);
}

// +bias -> final fp32 H (overwrites the GEMM output buffer, rows < N only)
__global__ void dgi_finish2(const float* __restrict__ acc, const float* __restrict__ b2,
                            float* __restrict__ H, int total) {
    int idx = blockIdx.x * 256 + threadIdx.x;
    if (idx >= total) return;
    H[idx] = acc[idx] + b2[idx & 127];
}

// ---------------------------------------------------------------------------
// Summary: column sums of H (block-local accumulation, ~25K atomics total),
// then s = sigmoid(sum/N), Ws = Wd[0] @ s.
// ---------------------------------------------------------------------------
__global__ void dgi_zero128(float* __restrict__ p) { p[threadIdx.x] = 0.0f; }

__global__ __launch_bounds__(256) void dgi_colsum(const float* __restrict__ H,
                                                  float* __restrict__ ssum, int n) {
    int f = threadIdx.x & 127;
    int half = threadIdx.x >> 7;   // 0/1
    int base = blockIdx.x * 512;
    float a = 0.0f;
    for (int r = half; r < 512; r += 2) {
        int i = base + r;
        if (i < n) a += H[(size_t)i * 128 + f];
    }
    __shared__ float tmp[128];
    if (half == 0) tmp[f] = a;
    __syncthreads();
    if (half == 1) unsafeAtomicAdd(&ssum[f], a + tmp[f]);
}

__global__ void dgi_summary(const float* __restrict__ ssum, const float* __restrict__ Wd,
                            float invN, float* __restrict__ Ws) {
    __shared__ float s[128];
    int t = threadIdx.x;
    s[t] = 1.0f / (1.0f + __expf(-ssum[t] * invN));
    __syncthreads();
    float a = 0.0f;
    for (int j = 0; j < 128; ++j) a += Wd[t * 128 + j] * s[j];
    Ws[t] = a;
}

// pos/neg: out[row] = dot(H[row], Ws) + bd  (one wave per row, shfl reduce)
__global__ __launch_bounds__(256) void dgi_dotrows(const float* __restrict__ H,
                                                   const float* __restrict__ Ws,
                                                   const float* __restrict__ bd,
                                                   float* __restrict__ out, int n) {
    int row = blockIdx.x * 8 + (threadIdx.x >> 5);
    if (row >= n) return;
    int lane = threadIdx.x & 31;
    float4 h = ((const float4*)(H + (size_t)row * 128))[lane];
    float4 w = ((const float4*)Ws)[lane];
    float v = h.x * w.x + h.y * w.y + h.z * w.z + h.w * w.w;
#pragma unroll
    for (int off = 16; off > 0; off >>= 1) v += __shfl_down(v, off, 32);
    if (lane == 0) out[row] = v + bd[0];
}

// ---------------------------------------------------------------------------
// Host launcher
// ---------------------------------------------------------------------------
extern "C" void kernel_launch(void* const* d_in, const int* in_sizes, int n_in,
                              void* d_out, int out_size, void* d_ws, size_t ws_size,
                              hipStream_t stream) {
    (void)n_in; (void)out_size; (void)ws_size;
    const float* x    = (const float*)d_in[0];
    const int*   ei   = (const int*)d_in[1];
    const int*   perm = (const int*)d_in[2];
    const float* W1   = (const float*)d_in[3];
    const float* b1   = (const float*)d_in[4];
    const float* W2   = (const float*)d_in[5];
    const float* b2   = (const float*)d_in[6];
    const float* Wd   = (const float*)d_in[7];
    const float* bd   = (const float*)d_in[8];

    const int N = in_sizes[0] / 512;
    const int E = in_sizes[1] / 2;
    const int Npad = ((N + 127) / 128) * 128;
    const int* src = ei;
    const int* dst = ei + E;

    char* wp = (char*)d_ws;
    auto carve = [&](size_t bytes) -> void* {
        void* r = (void*)wp;
        wp += (bytes + 255) & ~(size_t)255;
        return r;
    };
    float*          dinv = (float*)carve((size_t)Npad * 4);
    unsigned short* xb   = (unsigned short*)carve((size_t)Npad * 512 * 2);
    unsigned short* W1b  = (unsigned short*)carve((size_t)512 * 128 * 2);
    unsigned short* W2b  = (unsigned short*)carve((size_t)128 * 128 * 2);
    float*          Hlin = (float*)carve((size_t)Npad * 128 * 4);
    float*          acc  = (float*)carve((size_t)Npad * 128 * 4);
    unsigned short* h1b  = (unsigned short*)carve((size_t)Npad * 128 * 2);
    float*          ssum = (float*)carve(512);
    float*          Ws   = (float*)carve(512);

    const int totNF = N * 128;
    const int totPF = Npad * 128;

    // --- degrees / normalization (shared by both passes) ---
    dgi_deg_init<<<(N + 255) / 256, 256, 0, stream>>>(dinv, N);
    dgi_deg_count<<<(E + 255) / 256, 256, 0, stream>>>(dinv, dst, E);
    dgi_deg_fin<<<(N + 255) / 256, 256, 0, stream>>>(dinv, N);

    // --- pack weights to bf16 WMMA-B layout ---
    dgi_pack_w<<<(512 * 128 + 255) / 256, 256, 0, stream>>>(W1, W1b, 512);
    dgi_pack_w<<<(128 * 128 + 255) / 256, 256, 0, stream>>>(W2, W2b, 128);

    for (int pass = 0; pass < 2; ++pass) {
        const int* pm = pass ? perm : nullptr;
        // features -> bf16 (pad rows zeroed; corrupted pass gathers via perm)
        dgi_conv_x<<<(Npad * 512 + 255) / 256, 256, 0, stream>>>(x, pm, xb, N, Npad);
        // GCN layer 1
        dgi_gemm_bf16<<<Npad / 128, 256, 0, stream>>>(xb, W1b, Hlin, 512);
        dgi_selfinit<<<(totNF + 255) / 256, 256, 0, stream>>>(Hlin, dinv, acc, totNF);
        dgi_scatter<<<(E + 7) / 8, 256, 0, stream>>>(Hlin, acc, dinv, src, dst, E);
        dgi_finish1<<<(totPF + 255) / 256, 256, 0, stream>>>(acc, b1, h1b, N, Npad);
        // GCN layer 2
        dgi_gemm_bf16<<<Npad / 128, 256, 0, stream>>>(h1b, W2b, Hlin, 128);
        dgi_selfinit<<<(totNF + 255) / 256, 256, 0, stream>>>(Hlin, dinv, acc, totNF);
        dgi_scatter<<<(E + 7) / 8, 256, 0, stream>>>(Hlin, acc, dinv, src, dst, E);
        dgi_finish2<<<(totNF + 255) / 256, 256, 0, stream>>>(acc, b2, Hlin, totNF);

        if (pass == 0) {
            // summary from the clean embeddings only
            dgi_zero128<<<1, 128, 0, stream>>>(ssum);
            dgi_colsum<<<(N + 511) / 512, 256, 0, stream>>>(Hlin, ssum, N);
            dgi_summary<<<1, 128, 0, stream>>>(ssum, Wd, 1.0f / (float)N, Ws);
        }
        float* outp = (float*)d_out + (size_t)pass * N;   // pos then neg
        dgi_dotrows<<<(N + 7) / 8, 256, 0, stream>>>(Hlin, Ws, bd, outp, N);
    }
}
